// VanillaGCNEncoder_5806795784249
// MI455X (gfx1250) — matrix operand order repaired and checked by reading, not compiled, not run on hardware
//
#include <hip/hip_runtime.h>
#include <hip/hip_bf16.h>
#include <stdint.h>

// ---------------------------------------------------------------------------
// GCN forward on gfx1250 (MI455X): bf16 WMMA GEMMs + L2-resident edge scatter
// using native global_atomic_add_f32 (pinned via inline asm).
// ---------------------------------------------------------------------------

typedef __bf16 bf16_t;
typedef __attribute__((ext_vector_type(16))) __bf16 v16bf;
typedef __attribute__((ext_vector_type(8)))  float  v8f;

#define GCN_N_NODES 100000
#define GCN_N_EDGES 1600000

// Non-returning fp32 atomic add: guaranteed GLOBAL_ATOMIC_ADD_F32 (no CAS
// loop fallback). Tracked by STOREcnt; drained by S_ENDPGM's implicit
// wait-idle, so no explicit s_wait needed inside the kernel.
__device__ __forceinline__ void fatomic_add(float* p, float v) {
  asm volatile("global_atomic_add_f32 %0, %1, off"
               :
               : "v"(p), "v"(v)
               : "memory");
}

// ------------------------- elementwise helpers -----------------------------

__global__ void k_fill(float* __restrict__ p, float v, int n) {
  int i = blockIdx.x * blockDim.x + threadIdx.x;
  if (i < n) p[i] = v;
}

__global__ void k_deg_scatter(const long long* __restrict__ col,
                              const float* __restrict__ ew,
                              float* __restrict__ deg, int e_total) {
  int e = blockIdx.x * blockDim.x + threadIdx.x;
  if (e < e_total) fatomic_add(&deg[col[e]], ew[e]);
}

__global__ void k_rsqrt_inplace(float* __restrict__ p, int n) {
  int i = blockIdx.x * blockDim.x + threadIdx.x;
  if (i < n) p[i] = rsqrtf(p[i]);
}

__global__ void k_edge_norm(const long long* __restrict__ row,
                            const long long* __restrict__ col,
                            const float* __restrict__ ew,
                            const float* __restrict__ dis,
                            float* __restrict__ nrm, int e_total) {
  int e = blockIdx.x * blockDim.x + threadIdx.x;
  if (e < e_total) nrm[e] = dis[row[e]] * ew[e] * dis[col[e]];
}

__global__ void k_cvt_bf16(const float* __restrict__ src,
                           bf16_t* __restrict__ dst, int n) {
  int i = blockIdx.x * blockDim.x + threadIdx.x;
  if (i < n) dst[i] = (bf16_t)src[i];
}

// agg + dis^2 * xw + bias, ReLU, emit bf16 for next layer's WMMA input.
__global__ void k_combine_relu(const float* __restrict__ agg,
                               const float* __restrict__ xw,
                               const float* __restrict__ dis,
                               const float* __restrict__ bias,
                               bf16_t* __restrict__ out, int nrows) {
  int t = blockIdx.x * blockDim.x + threadIdx.x;
  if (t >= nrows * 64) return;
  int i = t >> 6;
  int c = t & 63;
  float d = dis[i];
  float v = agg[t] + d * d * xw[t] + bias[c];
  out[t] = (bf16_t)fmaxf(v, 0.0f);
}

// ------------------------- edge scatter (atomics) --------------------------
// 16 threads per edge, each: one coalesced float4 gather from the L2-resident
// xw table (25.6 MB << 192 MB L2) + 4 native f32 L2 atomics into agg.
__global__ void k_scatter(const long long* __restrict__ row,
                          const long long* __restrict__ col,
                          const float* __restrict__ nrm,
                          const float* __restrict__ xw,
                          float* __restrict__ agg, int e_total) {
  int t = blockIdx.x * blockDim.x + threadIdx.x;
  int e = t >> 4;
  if (e >= e_total) return;
  int j = t & 15;
  long long r = row[e];
  long long c = col[e];
  float nm = nrm[e];
  const float4 v = *(const float4*)(xw + ((size_t)r << 6) + (j << 2));
  float* dst = agg + ((size_t)c << 6) + (j << 2);
  fatomic_add(dst + 0, nm * v.x);
  fatomic_add(dst + 1, nm * v.y);
  fatomic_add(dst + 2, nm * v.z);
  fatomic_add(dst + 3, nm * v.w);
}

// ------------------------- WMMA fragment loaders ---------------------------
// A (16x32 bf16, MxK): lane = M (mod 16); VGPR q holds K pair at
//   pair_index = (q/4)*8 + (lane/16)*4 + (q%4), second K-half at +16 pairs.
// (Compiler coalesces these into two global_load_b128 per K-half.)
__device__ __forceinline__ void load_A64(const bf16_t* __restrict__ rowbase,
                                         int lane, v16bf& a0, v16bf& a1) {
  const unsigned* rowp = (const unsigned*)rowbase;  // 16 bf16 pairs per K-half
  int g = (lane >> 4) & 1;
  union { v16bf v; unsigned u[8]; } A0, A1;
#pragma unroll
  for (int q = 0; q < 8; ++q) {
    int p = ((q >> 2) << 3) + (g << 2) + (q & 3);
    A0.u[q] = rowp[p];        // K = 0..31
    A1.u[q] = rowp[16 + p];   // K = 32..63
  }
  a0 = A0.v;
  a1 = A1.v;
}

// B (32x16 bf16, KxN): lane = N (mod 16); element s holds
// K = (s/8)*16 + (lane/16)*8 + (s%8) (same K order as A), strided loads
// from a row-major [K x ldn] weight matrix.
__device__ __forceinline__ v16bf load_B(const bf16_t* __restrict__ W, int ldn,
                                        int kbase, int n0, int lane) {
  int n = n0 + (lane & 15);
  int g = (lane >> 4) & 1;
  union { v16bf v; bf16_t e[16]; } B;
#pragma unroll
  for (int s = 0; s < 16; ++s) {
    int k = kbase + ((s >> 3) << 4) + (g << 3) + (s & 7);
    B.e[s] = W[(size_t)k * ldn + n];
  }
  return B.v;
}

__device__ __forceinline__ v8f wmma_bf16(v16bf a, v16bf b, v8f c) {
  return __builtin_amdgcn_wmma_f32_16x16x32_bf16(
      /*neg_a=*/false, a, /*neg_b=*/false, b,
      /*c_mod=*/(short)0, c, /*reuse_a=*/false, /*reuse_b=*/false);
}

// ------------------------- GEMM: [nrows x 64] @ [64 x 64] ------------------
// One wave per 16-row tile: 4 n-tiles x 2 k-steps = 8 v_wmma per wave.
__global__ void __launch_bounds__(256)
k_gemm64(const bf16_t* __restrict__ Xb, const bf16_t* __restrict__ Wb,
         float* __restrict__ Y, int nrows) {
  int gwave = (blockIdx.x * blockDim.x + threadIdx.x) >> 5;
  int lane = threadIdx.x & 31;
  int m0 = gwave << 4;
  if (m0 >= nrows) return;  // wave-uniform: EXEC stays all-ones for WMMA

  int M = m0 + (lane & 15);
  v16bf a0, a1;
  load_A64(Xb + ((size_t)M << 6), lane, a0, a1);

  v8f acc0 = {}, acc1 = {}, acc2 = {}, acc3 = {};
  acc0 = wmma_bf16(a0, load_B(Wb, 64, 0, 0, lane), acc0);
  acc0 = wmma_bf16(a1, load_B(Wb, 64, 32, 0, lane), acc0);
  acc1 = wmma_bf16(a0, load_B(Wb, 64, 0, 16, lane), acc1);
  acc1 = wmma_bf16(a1, load_B(Wb, 64, 32, 16, lane), acc1);
  acc2 = wmma_bf16(a0, load_B(Wb, 64, 0, 32, lane), acc2);
  acc2 = wmma_bf16(a1, load_B(Wb, 64, 32, 32, lane), acc2);
  acc3 = wmma_bf16(a0, load_B(Wb, 64, 0, 48, lane), acc3);
  acc3 = wmma_bf16(a1, load_B(Wb, 64, 32, 48, lane), acc3);

  // D layout: VGPR r -> row m0 + r + 8*(lane/16); lane%16 -> column.
  int mrow = m0 + ((lane >> 4) << 3);
  int n = lane & 15;
#pragma unroll
  for (int r = 0; r < 8; ++r) {
    size_t base = ((size_t)(mrow + r) << 6) + n;
    Y[base + 0]  = acc0[r];
    Y[base + 16] = acc1[r];
    Y[base + 32] = acc2[r];
    Y[base + 48] = acc3[r];
  }
}

// --------------- Final layer: tanh([nrows x 64] @ [64 x 32] + b) -----------
__global__ void __launch_bounds__(256)
k_gemm_out(const bf16_t* __restrict__ Xb, const bf16_t* __restrict__ Wb,
           const float* __restrict__ bias, float* __restrict__ Y, int nrows) {
  int gwave = (blockIdx.x * blockDim.x + threadIdx.x) >> 5;
  int lane = threadIdx.x & 31;
  int m0 = gwave << 4;
  if (m0 >= nrows) return;

  int M = m0 + (lane & 15);
  v16bf a0, a1;
  load_A64(Xb + ((size_t)M << 6), lane, a0, a1);

  v8f acc0 = {}, acc1 = {};
  acc0 = wmma_bf16(a0, load_B(Wb, 32, 0, 0, lane), acc0);
  acc0 = wmma_bf16(a1, load_B(Wb, 32, 32, 0, lane), acc0);
  acc1 = wmma_bf16(a0, load_B(Wb, 32, 0, 16, lane), acc1);
  acc1 = wmma_bf16(a1, load_B(Wb, 32, 32, 16, lane), acc1);

  int mrow = m0 + ((lane >> 4) << 3);
  int n = lane & 15;
  float bi0 = bias[n];
  float bi1 = bias[16 + n];
#pragma unroll
  for (int r = 0; r < 8; ++r) {
    size_t base = ((size_t)(mrow + r) << 5) + n;
    Y[base + 0]  = tanhf(acc0[r] + bi0);
    Y[base + 16] = tanhf(acc1[r] + bi1);
  }
}

// ---------------------------------------------------------------------------

extern "C" void kernel_launch(void* const* d_in, const int* in_sizes, int n_in,
                              void* d_out, int out_size, void* d_ws, size_t ws_size,
                              hipStream_t stream) {
  (void)in_sizes; (void)n_in; (void)out_size; (void)ws_size;

  const float*     x    = (const float*)d_in[0];
  const long long* ei   = (const long long*)d_in[1];  // int64 edge_index [2,E]
  const float*     ew   = (const float*)d_in[2];
  const float*     W1   = (const float*)d_in[3];
  const float*     b1   = (const float*)d_in[4];
  const float*     W2   = (const float*)d_in[5];
  const float*     b2   = (const float*)d_in[6];
  const float*     Wout = (const float*)d_in[7];
  const float*     bout = (const float*)d_in[8];

  const long long* row = ei;
  const long long* col = ei + GCN_N_EDGES;

  // Workspace carve-up (256B aligned), ~71 MB total.
  char* ws = (char*)d_ws;
  size_t off = 0;
  auto alloc = [&](size_t bytes) -> void* {
    off = (off + 255) & ~(size_t)255;
    void* p = ws + off;
    off += bytes;
    return p;
  };
  float*  dis = (float*)alloc((size_t)GCN_N_NODES * 4);            // deg -> rsqrt in place
  float*  nrm = (float*)alloc((size_t)GCN_N_EDGES * 4);
  bf16_t* inb = (bf16_t*)alloc((size_t)GCN_N_NODES * 64 * 2);      // current layer input (bf16)
  float*  xw  = (float*)alloc((size_t)GCN_N_NODES * 64 * 4);
  float*  agg = (float*)alloc((size_t)GCN_N_NODES * 64 * 4);
  bf16_t* w1b = (bf16_t*)alloc(64 * 64 * 2);
  bf16_t* w2b = (bf16_t*)alloc(64 * 64 * 2);
  bf16_t* wob = (bf16_t*)alloc(64 * 32 * 2);

  const int T = 256;
  const int NB   = (GCN_N_NODES + T - 1) / T;
  const int EB   = (GCN_N_EDGES + T - 1) / T;
  const int NC   = (GCN_N_NODES * 64 + T - 1) / T;          // 6.4M elems
  const int SCAT = (GCN_N_EDGES * 16 + T - 1) / T;          // 25.6M threads

  // degree / normalization (shared by both conv layers)
  k_fill<<<NB, T, 0, stream>>>(dis, 1.0f, GCN_N_NODES);     // self-loop weight 1
  k_deg_scatter<<<EB, T, 0, stream>>>(col, ew, dis, GCN_N_EDGES);
  k_rsqrt_inplace<<<NB, T, 0, stream>>>(dis, GCN_N_NODES);
  k_edge_norm<<<EB, T, 0, stream>>>(row, col, ew, dis, nrm, GCN_N_EDGES);

  // weights and input to bf16 (WMMA operand precision; f32 accumulation)
  k_cvt_bf16<<<(64 * 64 + T - 1) / T, T, 0, stream>>>(W1, w1b, 64 * 64);
  k_cvt_bf16<<<(64 * 64 + T - 1) / T, T, 0, stream>>>(W2, w2b, 64 * 64);
  k_cvt_bf16<<<(64 * 32 + T - 1) / T, T, 0, stream>>>(Wout, wob, 64 * 32);
  k_cvt_bf16<<<NC, T, 0, stream>>>(x, inb, GCN_N_NODES * 64);

  const int nwaves  = (GCN_N_NODES + 15) / 16;
  const int gblocks = (nwaves * 32 + T - 1) / T;

  // ---- layer 1 ----
  k_gemm64<<<gblocks, T, 0, stream>>>(inb, w1b, xw, GCN_N_NODES);
  k_fill<<<NC, T, 0, stream>>>(agg, 0.0f, GCN_N_NODES * 64);
  k_scatter<<<SCAT, T, 0, stream>>>(row, col, nrm, xw, agg, GCN_N_EDGES);
  k_combine_relu<<<NC, T, 0, stream>>>(agg, xw, dis, b1, inb, GCN_N_NODES);

  // ---- layer 2 ----
  k_gemm64<<<gblocks, T, 0, stream>>>(inb, w2b, xw, GCN_N_NODES);
  k_fill<<<NC, T, 0, stream>>>(agg, 0.0f, GCN_N_NODES * 64);
  k_scatter<<<SCAT, T, 0, stream>>>(row, col, nrm, xw, agg, GCN_N_EDGES);
  k_combine_relu<<<NC, T, 0, stream>>>(agg, xw, dis, b2, inb, GCN_N_NODES);

  // ---- output projection + tanh ----
  k_gemm_out<<<gblocks, T, 0, stream>>>(inb, wob, bout, (float*)d_out, GCN_N_NODES);
}
